// MultiLoRALinear_31421980737730
// MI455X (gfx1250) — matrix operand order repaired and checked by reading, not compiled, not run on hardware
//
#include <hip/hip_runtime.h>

#define T_DIM  16384
#define IN_DIM 4096
#define OUT_DIM 4096
#define R_DIM  16
#define E_DIM  8

typedef __attribute__((ext_vector_type(16))) __bf16 bf16x16;
typedef __attribute__((ext_vector_type(2)))  __bf16 bf16x2;
typedef __attribute__((ext_vector_type(2)))  float  float2v;
typedef __attribute__((ext_vector_type(8)))  float  floatx8;

union Frag {
    uint4        q[2];
    bf16x16      v;
    unsigned int u[8];
};

// f32 pair -> packed bf16 pair; lowers to v_cvt_pk_bf16_f32 (verified round 2)
__device__ __forceinline__ unsigned int pack2(float lo, float hi) {
    float2v f;
    f[0] = lo; f[1] = hi;
    union { bf16x2 h; unsigned int u; } c;
    c.h = __builtin_convertvector(f, bf16x2);
    return c.u;
}
__device__ __forceinline__ unsigned short f2bf(float f) {
    return (unsigned short)(pack2(f, 0.0f) & 0xFFFFu);
}
__device__ __forceinline__ floatx8 zero8() {
    floatx8 z;
#pragma unroll
    for (int i = 0; i < 8; ++i) z[i] = 0.0f;
    return z;
}

// ---------------------------------------------------------------------------
// Kernel 1: S[t, r] = sum_k x[t,k] * lora_A[e(t)][k][r]  (bf16 result in d_ws)
// One wave per 16-row tile; WMMA 16x16x32 bf16, f32 accumulate.
// ---------------------------------------------------------------------------
__global__ __launch_bounds__(256) void lora_s_kernel(
    const float* __restrict__ x, const int* __restrict__ offsets,
    const float* __restrict__ lora_A, unsigned short* __restrict__ S) {
    const int lane = threadIdx.x & 31;
    const int wave = threadIdx.x >> 5;
    const int row_base = blockIdx.x * 128 + wave * 16;
    const int half = lane >> 4;   // 0: lanes 0-15, 1: lanes 16-31
    const int l15  = lane & 15;

    // adapter id = #(offsets <= row) (searchsorted right); groups 2048-aligned
    int e = 0;
#pragma unroll
    for (int j = 0; j < E_DIM; ++j) e += (row_base >= offsets[j]) ? 1 : 0;
    const float* Ae = lora_A + (size_t)e * IN_DIM * R_DIM;

    floatx8 acc = zero8();
    const int arow = row_base + l15;

    for (int k0 = 0; k0 < IN_DIM; k0 += 32) {
        // A fragment (x rows): lanes 0-15 hold K=0..15; lanes 16-31 K=16..31
        Frag a;
        const float* xp = x + (size_t)arow * IN_DIM + k0 + half * 16;
        float4 f0 = *(const float4*)(xp + 0);
        float4 f1 = *(const float4*)(xp + 4);
        float4 f2 = *(const float4*)(xp + 8);
        float4 f3 = *(const float4*)(xp + 12);
        a.u[0] = pack2(f0.x, f0.y); a.u[1] = pack2(f0.z, f0.w);
        a.u[2] = pack2(f1.x, f1.y); a.u[3] = pack2(f1.z, f1.w);
        a.u[4] = pack2(f2.x, f2.y); a.u[5] = pack2(f2.z, f2.w);
        a.u[6] = pack2(f3.x, f3.y); a.u[7] = pack2(f3.z, f3.w);
        // B fragment (lora_A columns): lane n=l15 holds column n; lanes 0-15 K=0..15,
        // lanes 16-31 K=16..31, K ascending packed 2 per VGPR.
        Frag b;
        const float* bp = Ae + (size_t)(k0 + half * 16) * R_DIM + l15;
#pragma unroll
        for (int j = 0; j < 8; ++j)
            b.u[j] = pack2(bp[(size_t)(2 * j) * R_DIM], bp[(size_t)(2 * j + 1) * R_DIM]);
        acc = __builtin_amdgcn_wmma_f32_16x16x32_bf16(false, a.v, false, b.v,
                                                      (short)0, acc, false, false);
    }
    // C layout: VGPR v, lanes 0-15 -> M=v; lanes 16-31 -> M=8+v; N=lane&15
#pragma unroll
    for (int v = 0; v < 8; ++v) {
        int row = row_base + v + 8 * half;
        S[(size_t)row * R_DIM + l15] = f2bf(acc[v]);
    }
}

// ---------------------------------------------------------------------------
// Kernel 2: out = x @ W^T + b + S @ lora_B[e]
// 128(M) x 256(N) block tile, 8 waves (2x4), wave tile 64x64 (4x4 WMMA accs).
// Double-buffered LDS bf16 tiles, one barrier per K-step, pipelined globals.
// ---------------------------------------------------------------------------
__global__ __launch_bounds__(256) void lora_main_kernel(
    const float* __restrict__ x, const int* __restrict__ offsets,
    const float* __restrict__ W, const float* __restrict__ bias,
    const float* __restrict__ lora_B, const unsigned short* __restrict__ S,
    float* __restrict__ out) {
    // row stride 40 bf16 (80B) to spread LDS banks; 16B aligned chunks
    __shared__ __align__(16) unsigned short xs[2][128 * 40];   // 2 x 10 KB
    __shared__ __align__(16) unsigned short wsh[2][256 * 40];  // 2 x 20 KB

    const int tid  = threadIdx.x;
    const int lane = tid & 31;
    const int wave = tid >> 5;
    const int wm   = wave >> 2;   // 0..1 : 64-row half
    const int wn   = wave & 3;    // 0..3 : 64-col quarter
    const int half = lane >> 4;
    const int l15  = lane & 15;

    const int bm = blockIdx.y * 128;
    const int bn = blockIdx.x * 256;

    // x staging: thread -> (row = tid>>1, 16-col half = tid&1); 16 f32 each
    const int srow = tid >> 1;
    const int scw  = tid & 1;
    const float* xg = x + (size_t)(bm + srow) * IN_DIM + scw * 16;
    const int sidx = srow * 40 + scw * 16;
    // W staging: thread -> full row tid; 32 f32 each
    const float* wg = W + (size_t)(bn + tid) * IN_DIM;
    const int widx = tid * 40;

    floatx8 acc[4][4];
#pragma unroll
    for (int i = 0; i < 4; ++i)
#pragma unroll
        for (int j = 0; j < 4; ++j) acc[i][j] = zero8();

    float4 rx0, rx1, rx2, rx3;
    float4 rw0, rw1, rw2, rw3, rw4, rw5, rw6, rw7;

    auto loadStage = [&](int k0) {
        rx0 = *(const float4*)(xg + k0);      rx1 = *(const float4*)(xg + k0 + 4);
        rx2 = *(const float4*)(xg + k0 + 8);  rx3 = *(const float4*)(xg + k0 + 12);
        rw0 = *(const float4*)(wg + k0);      rw1 = *(const float4*)(wg + k0 + 4);
        rw2 = *(const float4*)(wg + k0 + 8);  rw3 = *(const float4*)(wg + k0 + 12);
        rw4 = *(const float4*)(wg + k0 + 16); rw5 = *(const float4*)(wg + k0 + 20);
        rw6 = *(const float4*)(wg + k0 + 24); rw7 = *(const float4*)(wg + k0 + 28);
    };
    auto storeStage = [&](int buf) {
        *(uint4*)&xs[buf][sidx]      = make_uint4(pack2(rx0.x, rx0.y), pack2(rx0.z, rx0.w),
                                                  pack2(rx1.x, rx1.y), pack2(rx1.z, rx1.w));
        *(uint4*)&xs[buf][sidx + 8]  = make_uint4(pack2(rx2.x, rx2.y), pack2(rx2.z, rx2.w),
                                                  pack2(rx3.x, rx3.y), pack2(rx3.z, rx3.w));
        *(uint4*)&wsh[buf][widx]      = make_uint4(pack2(rw0.x, rw0.y), pack2(rw0.z, rw0.w),
                                                   pack2(rw1.x, rw1.y), pack2(rw1.z, rw1.w));
        *(uint4*)&wsh[buf][widx + 8]  = make_uint4(pack2(rw2.x, rw2.y), pack2(rw2.z, rw2.w),
                                                   pack2(rw3.x, rw3.y), pack2(rw3.z, rw3.w));
        *(uint4*)&wsh[buf][widx + 16] = make_uint4(pack2(rw4.x, rw4.y), pack2(rw4.z, rw4.w),
                                                   pack2(rw5.x, rw5.y), pack2(rw5.z, rw5.w));
        *(uint4*)&wsh[buf][widx + 24] = make_uint4(pack2(rw6.x, rw6.y), pack2(rw6.z, rw6.w),
                                                   pack2(rw7.x, rw7.y), pack2(rw7.z, rw7.w));
    };

    constexpr int NK = IN_DIM / 32;
    loadStage(0);
    storeStage(0);

    for (int kt = 0; kt < NK; ++kt) {
        const int cur = kt & 1;
        __syncthreads();               // buf[cur] fully written; buf[cur^1] readers done
        if (kt + 1 < NK) loadStage((kt + 1) * 32);   // in flight during WMMA below

        Frag afr[4], bfr[4];
#pragma unroll
        for (int mt = 0; mt < 4; ++mt) {  // A: lanes 0-15 K=0..15; 16-31 K=16..31
            int r = wm * 64 + mt * 16 + l15;
            int base = r * 40 + half * 16;
            afr[mt].q[0] = *(const uint4*)&xs[cur][base];
            afr[mt].q[1] = *(const uint4*)&xs[cur][base + 8];
        }
#pragma unroll
        for (int nt = 0; nt < 4; ++nt) {  // B: lane = column (= W row), 32B contiguous K
            int c = wn * 64 + nt * 16 + l15;
            int base = c * 40 + half * 16;
            bfr[nt].q[0] = *(const uint4*)&wsh[cur][base];
            bfr[nt].q[1] = *(const uint4*)&wsh[cur][base + 8];
        }
#pragma unroll
        for (int mt = 0; mt < 4; ++mt)
#pragma unroll
            for (int nt = 0; nt < 4; ++nt)
                acc[mt][nt] = __builtin_amdgcn_wmma_f32_16x16x32_bf16(
                    false, afr[mt].v, false, bfr[nt].v, (short)0, acc[mt][nt], false, false);

        if (kt + 1 < NK) storeStage(cur ^ 1);  // fill other buffer for next iteration
    }

    // ---- fused epilogue: rank-16 LoRA via one K-padded WMMA per tile ----
    int e = 0;
#pragma unroll
    for (int j = 0; j < E_DIM; ++j) e += (bm >= offsets[j]) ? 1 : 0;
    const float* Be = lora_B + (size_t)e * R_DIM * OUT_DIM;

    Frag sB[4];
#pragma unroll
    for (int nt = 0; nt < 4; ++nt) {
        if (half == 0) {  // K=0..15 real data: column of lora_B[e]
            const float* bp = Be + bn + wn * 64 + nt * 16 + l15;
#pragma unroll
            for (int j = 0; j < 8; ++j)
                sB[nt].u[j] = pack2(bp[(size_t)(2 * j) * OUT_DIM],
                                    bp[(size_t)(2 * j + 1) * OUT_DIM]);
        } else {          // K=16..31 zero pad
#pragma unroll
            for (int j = 0; j < 8; ++j) sB[nt].u[j] = 0u;
        }
    }
#pragma unroll
    for (int mt = 0; mt < 4; ++mt) {
        Frag sA;
        int row = bm + wm * 64 + mt * 16 + l15;
        sA.q[0] = half == 0 ? *(const uint4*)&S[(size_t)row * R_DIM] : make_uint4(0u, 0u, 0u, 0u);      // K 0..7 / 16..23 pad
        sA.q[1] = half == 0 ? *(const uint4*)&S[(size_t)row * R_DIM + 8] : make_uint4(0u, 0u, 0u, 0u);  // K 8..15 / 24..31 pad
#pragma unroll
        for (int nt = 0; nt < 4; ++nt)
            acc[mt][nt] = __builtin_amdgcn_wmma_f32_16x16x32_bf16(
                false, sA.v, false, sB[nt].v, (short)0, acc[mt][nt], false, false);
    }

    // ---- bias + store ----
#pragma unroll
    for (int nt = 0; nt < 4; ++nt) {
        int col = bn + wn * 64 + nt * 16 + l15;
        float bv = bias[col];
#pragma unroll
        for (int mt = 0; mt < 4; ++mt) {
            int row0 = bm + wm * 64 + mt * 16 + half * 8;
#pragma unroll
            for (int v = 0; v < 8; ++v)
                out[(size_t)(row0 + v) * OUT_DIM + col] = acc[mt][nt][v] + bv;
        }
    }
}

extern "C" void kernel_launch(void* const* d_in, const int* in_sizes, int n_in,
                              void* d_out, int out_size, void* d_ws, size_t ws_size,
                              hipStream_t stream) {
    const float* x       = (const float*)d_in[0];
    const int*   offsets = (const int*)d_in[1];
    const float* W       = (const float*)d_in[2];
    const float* b       = (const float*)d_in[3];
    const float* lora_A  = (const float*)d_in[4];
    const float* lora_B  = (const float*)d_in[5];
    float*       out     = (float*)d_out;
    unsigned short* S    = (unsigned short*)d_ws;  // T*R bf16 = 512 KB scratch

    lora_s_kernel<<<dim3(T_DIM / 128), dim3(256), 0, stream>>>(x, offsets, lora_A, S);
    lora_main_kernel<<<dim3(OUT_DIM / 256, T_DIM / 128), dim3(256), 0, stream>>>(
        x, offsets, W, b, lora_B, S, out);
}
